// ALALLaDA_33767032881178
// MI455X (gfx1250) — matrix operand order repaired and checked by reading, not compiled, not run on hardware
//
#include <hip/hip_runtime.h>

#define ALPHA   0.08f
#define EPS_LN  1e-5f
#define B_      2
#define S_      2048
#define D_      2048
#define K_      8
#define FH_     512
#define RANGE_  8
#define NTOK    (B_ * S_)

// ---- WMMA / TDM vector types (CDNA5 / gfx1250, wave32) ----
typedef __attribute__((ext_vector_type(16))) __bf16       v16bf;
typedef __attribute__((ext_vector_type(8)))  float        v8f;
typedef __attribute__((ext_vector_type(4))) unsigned int  v4u;
typedef __attribute__((ext_vector_type(4))) unsigned int  u32x4;
typedef __attribute__((ext_vector_type(8)))  int          i32x8;
typedef __attribute__((ext_vector_type(4)))  int          i32x4;

union Frag { v16bf v; v4u q[2]; };

__device__ inline unsigned short f2bf(float f) {
    union { float f; unsigned int u; } x; x.f = f;
    unsigned int u = x.u;
    u += 0x7FFFu + ((u >> 16) & 1u);         // round-to-nearest-even
    return (unsigned short)(u >> 16);
}
__device__ inline float bf2f(unsigned short h) {
    union { unsigned int u; float f; } x; x.u = ((unsigned int)h) << 16;
    return x.f;
}

// ---------------------------------------------------------------------------
// TDM 1-D copy: n8 x 8-byte elements, contiguous global -> contiguous LDS.
// Descriptor per CDNA5 ISA (D# group0/group1), type=2.
// This toolchain exposes the 6-arg builtin:
//   (u32x4 g0, i32x8 g1, i32x4 g2, i32x4 g3, i32x8 extra, i32 cpol)
// ---------------------------------------------------------------------------
__device__ inline void tdm_copy_1d(unsigned int lds_byte_off,
                                   unsigned long long gaddr, unsigned int n8) {
    u32x4 g0;
    g0.x = 1u;                                           // count=1 (valid D#)
    g0.y = lds_byte_off;                                 // lds_addr
    g0.z = (unsigned int)(gaddr & 0xFFFFFFFFull);        // global_addr[31:0]
    g0.w = (unsigned int)((gaddr >> 32) & 0x1FFFFFFull)  // global_addr[56:32]
         | (2u << 30);                                   // type=2 ("image")
    i32x8 g1;
    g1[0] = (int)(3u << 16);                             // data_size=3 (8B)
    g1[1] = (int)((n8 & 0xFFFFu) << 16);                 // tensor_dim0[15:0] @ [63:48]
    g1[2] = (int)(((n8 >> 16) & 0xFFFFu) | (1u << 16));  // tensor_dim0[31:16], tensor_dim1=1
    g1[3] = (int)((n8 & 0xFFFFu) << 16);                 // tile_dim0 @ [127:112]
    g1[4] = 0;                                           // tile_dim1=0, tile_dim2=0
    g1[5] = (int)n8;                                     // tensor_dim0_stride lo32
    g1[6] = 0;
    g1[7] = 0;
    i32x4 z4 = {0, 0, 0, 0};
    i32x8 z8 = {0, 0, 0, 0, 0, 0, 0, 0};
    __builtin_amdgcn_tensor_load_to_lds(g0, g1, z4, z4, z8, 0);
}

// A fragment: 16x32 bf16 row-major, k-contiguous rows (stride 40 elems).
__device__ inline v16bf load_fragA(const unsigned short* rowp, int lane) {
    const unsigned short* p = rowp + ((lane & 16) ? 8 : 0);
    Frag f;
    f.q[0] = *(const v4u*)(p);        // ds_load_b128
    f.q[1] = *(const v4u*)(p + 16);
    return f.v;
}
// B fragment: 32x16 bf16, LDS n-major, 32 contiguous K per n-row (stride 40).
__device__ inline v16bf load_fragB(const unsigned short* bbase, int ncol, int lane) {
    const unsigned short* p = bbase + ncol * 40 + ((lane & 16) ? 8 : 0);
    Frag f;
    f.q[0] = *(const v4u*)(p);
    f.q[1] = *(const v4u*)(p + 16);
    return f.v;
}

// ---------------------------------------------------------------------------
// Prep kernels: one-time fp32 -> bf16 conversion into TDM-ready tile images.
// Record = 40 bf16 (32 data + 8 pad), rows 16B aligned, chunk = contiguous.
// ---------------------------------------------------------------------------
__global__ __launch_bounds__(256) void prep_h_kernel(
    const float* __restrict__ hL, unsigned short* __restrict__ hp)
{
    int r = blockIdx.x * 256 + threadIdx.x;        // [0, 128*64*32)
    int row  = r & 31;
    int cc   = (r >> 5) & 63;
    int tile = r >> 11;
    const float* src = hL + (size_t)(tile * 32 + row) * D_ + cc * 32;
    unsigned short* dst = hp + (size_t)r * 40;
#pragma unroll
    for (int kk = 0; kk < 32; ++kk) dst[kk] = f2bf(src[kk]);
}

__global__ __launch_bounds__(256) void prep_w1_kernel(
    const float* __restrict__ W1, unsigned short* __restrict__ w1p)
{
    int r = blockIdx.x * 256 + threadIdx.x;        // [0, 8*64*512)
    int n  = r & 511;
    int cc = (r >> 9) & 63;
    int k  = r >> 15;
    unsigned short* dst = w1p + (size_t)r * 40;
#pragma unroll
    for (int kk = 0; kk < 32; ++kk)
        dst[kk] = f2bf(W1[((size_t)k * D_ + cc * 32 + kk) * FH_ + n]);
}

__global__ __launch_bounds__(256) void prep_w2_kernel(
    const float* __restrict__ W2, unsigned short* __restrict__ w2p)
{
    int r = blockIdx.x * 256 + threadIdx.x;        // [0, 8*4*16*512)
    int n  = r & 511;
    int cb = (r >> 9) & 15;
    int c2 = (r >> 13) & 3;
    int k  = r >> 15;
    unsigned short* dst = w2p + (size_t)r * 40;
#pragma unroll
    for (int kk = 0; kk < 32; ++kk)
        dst[kk] = f2bf(W2[((size_t)k * FH_ + cb * 32 + kk) * D_ + c2 * 512 + n]);
}

// ---------------------------------------------------------------------------
// Kernel: router  w[t,k] = softmax_k(h[t]·Wr[k] + br[k])
// ---------------------------------------------------------------------------
__global__ __launch_bounds__(256) void router_kernel(
    const float* __restrict__ hL, const float* __restrict__ Wr,
    const float* __restrict__ br, float* __restrict__ wout)
{
    __shared__ float logits[K_];
    const int t    = blockIdx.x;
    const int lane = threadIdx.x & 31;
    const int wave = threadIdx.x >> 5;
    const float* h  = hL + (size_t)t * D_;
    const float* wr = Wr + (size_t)wave * D_;
    float s = 0.f;
    for (int d = lane; d < D_; d += 32) s += h[d] * wr[d];
#pragma unroll
    for (int off = 16; off > 0; off >>= 1) s += __shfl_xor(s, off, 32);
    if (lane == 0) logits[wave] = s + br[wave];
    __syncthreads();
    if (threadIdx.x == 0) {
        float m = logits[0];
#pragma unroll
        for (int i = 1; i < K_; ++i) m = fmaxf(m, logits[i]);
        float e[K_]; float den = 0.f;
#pragma unroll
        for (int i = 0; i < K_; ++i) { e[i] = __expf(logits[i] - m); den += e[i]; }
        float inv = 1.f / den;
#pragma unroll
        for (int i = 0; i < K_; ++i) wout[(size_t)t * K_ + i] = e[i] * inv;
    }
}

// ---------------------------------------------------------------------------
// Main GEMM kernel: fused per-expert MLP, bf16 WMMA + TDM double buffering.
//   M[t,k,:] = GELU(h[t]·W1[k] + b1[k]) · W2[k] + b2[k]     (stored bf16)
// Block = (32-token tile, expert k), 256 threads = 8 waves, 16x128 acc/wave.
// LDS (bytes): A bufs 2x2560 @0, B bufs 2x40960 @5120, act 33792 @87040.
// ---------------------------------------------------------------------------
#define LDSB_A(buf)     ((buf) * 2560u)
#define LDSB_B(buf)     (5120u + (buf) * 40960u)
#define LDSB_A2         87040u
#define LDS_TOTAL_BYTES 120832u

__global__ __launch_bounds__(256) void expert_mlp_kernel(
    const unsigned short* __restrict__ hp,
    const unsigned short* __restrict__ w1p,
    const unsigned short* __restrict__ w2p,
    const float* __restrict__ b1, const float* __restrict__ b2,
    unsigned short* __restrict__ Mout)
{
    extern __shared__ __align__(16) unsigned short smem[];
    unsigned short* ldsA  = smem;                 // 2 bufs, 1280 elems each
    unsigned short* ldsB  = smem + 2560;          // 2 bufs, 20480 elems each
    unsigned short* ldsA2 = smem + 43520;         // 32 x 528 elems

    const int tid  = threadIdx.x;
    const int lane = tid & 31;
    const int wave = tid >> 5;
    const int k    = blockIdx.y;                  // expert
    const int tile = blockIdx.x;                  // 32-token tile
    const int tok0 = tile * 32;

    const int r0 = (wave >> 2) * 16;              // {0,16}
    const int c0 = (wave & 3) * 128;              // {0,128,256,384}

    const unsigned long long hpA  =
        (unsigned long long)(uintptr_t)(hp  + (size_t)tile * 64 * 32 * 40);
    const unsigned long long w1pB =
        (unsigned long long)(uintptr_t)(w1p + (size_t)k * 64 * 512 * 40);
    const unsigned long long w2pB =
        (unsigned long long)(uintptr_t)(w2p + (size_t)k * 4 * 16 * 512 * 40);

    v8f zero = {};
    v8f acc[8];
#pragma unroll
    for (int t = 0; t < 8; ++t) acc[t] = zero;

    // ---------------- Stage 1: acc[32x512] = h_tile[32x2048] x W1[k] --------
    if (wave == 0) {
        tdm_copy_1d(LDSB_A(0), hpA, 320);                 // 2560 B
        tdm_copy_1d(LDSB_B(0), w1pB, 5120);               // 40960 B
        __builtin_amdgcn_s_wait_tensorcnt((short)0);
    }
    __syncthreads();

    for (int c = 0; c < 64; ++c) {
        const int cur = c & 1, nxt = cur ^ 1;
        if (wave == 0 && c + 1 < 64) {                    // prefetch next chunk
            tdm_copy_1d(LDSB_A(nxt), hpA  + (size_t)(c + 1) * 2560u,  320);
            tdm_copy_1d(LDSB_B(nxt), w1pB + (size_t)(c + 1) * 40960u, 5120);
        }
        const unsigned short* aBase = ldsA + cur * 1280;
        const unsigned short* bBase = ldsB + cur * 20480;
        v16bf aF = load_fragA(aBase + (r0 + (lane & 15)) * 40, lane);
#pragma unroll
        for (int nt = 0; nt < 8; ++nt) {
            v16bf bF = load_fragB(bBase, c0 + nt * 16 + (lane & 15), lane);
            acc[nt] = __builtin_amdgcn_wmma_f32_16x16x32_bf16(
                false, aF, false, bF, (short)0, acc[nt], false, false);
        }
        if (wave == 0) __builtin_amdgcn_s_wait_tensorcnt((short)0);
        __syncthreads();
    }

    // bias + exact GELU -> bf16 activations in ldsA2[row][fh] (stride 528)
#pragma unroll
    for (int nt = 0; nt < 8; ++nt) {
#pragma unroll
        for (int i = 0; i < 8; ++i) {
            int row = r0 + ((lane & 16) ? (i + 8) : i);
            int col = c0 + nt * 16 + (lane & 15);
            float x = acc[nt][i] + b1[k * FH_ + col];
            float g = 0.5f * x * (1.0f + erff(x * 0.70710678118f));
            ldsA2[row * 528 + col] = f2bf(g);
        }
    }
    __syncthreads();

    // ---------------- Stage 2: M[32x2048] = act[32x512] x W2[k] -------------
    for (int c2 = 0; c2 < 4; ++c2) {
#pragma unroll
        for (int t = 0; t < 8; ++t) acc[t] = zero;
        const unsigned long long w2c = w2pB + (size_t)c2 * 16 * 40960u;
        if (wave == 0) {
            tdm_copy_1d(LDSB_B(0), w2c, 5120);
            __builtin_amdgcn_s_wait_tensorcnt((short)0);
        }
        __syncthreads();
        for (int cb = 0; cb < 16; ++cb) {
            const int cur = cb & 1, nxt = cur ^ 1;
            if (wave == 0 && cb + 1 < 16)
                tdm_copy_1d(LDSB_B(nxt), w2c + (size_t)(cb + 1) * 40960u, 5120);
            const unsigned short* bBase = ldsB + cur * 20480;
            v16bf aF = load_fragA(ldsA2 + (r0 + (lane & 15)) * 528 + cb * 32, lane);
#pragma unroll
            for (int nt = 0; nt < 8; ++nt) {
                v16bf bF = load_fragB(bBase, c0 + nt * 16 + (lane & 15), lane);
                acc[nt] = __builtin_amdgcn_wmma_f32_16x16x32_bf16(
                    false, aF, false, bF, (short)0, acc[nt], false, false);
            }
            if (wave == 0) __builtin_amdgcn_s_wait_tensorcnt((short)0);
            __syncthreads();
        }
        // + bias, store bf16 into M[t, k, d]
#pragma unroll
        for (int nt = 0; nt < 8; ++nt) {
#pragma unroll
            for (int i = 0; i < 8; ++i) {
                int row = r0 + ((lane & 16) ? (i + 8) : i);
                int col = c2 * 512 + c0 + nt * 16 + (lane & 15);
                float x = acc[nt][i] + b2[k * D_ + col];
                Mout[((size_t)(tok0 + row) * K_ + k) * D_ + col] = f2bf(x);
            }
        }
    }
}

// ---------------------------------------------------------------------------
// Kernel: neighbor blend + LayerNorm + masked delta.
// ---------------------------------------------------------------------------
__global__ __launch_bounds__(256) void blend_kernel(
    const float* __restrict__ hL, const int* __restrict__ mask,
    const float* __restrict__ wbuf, const unsigned short* __restrict__ Mbuf,
    float* __restrict__ out)
{
    __shared__ float red[16];
    __shared__ float wsh[K_];
    const int a = blockIdx.x;
    const int b = a / S_, s = a % S_;
    const int tid = threadIdx.x, lane = tid & 31, wave = tid >> 5;

    if (tid < K_) wsh[tid] = wbuf[(size_t)a * K_ + tid];
    __syncthreads();

    float acc[8];
#pragma unroll
    for (int j = 0; j < 8; ++j) acc[j] = 0.f;
    float cnt = 0.f;

    for (int o = -RANGE_; o <= RANGE_; ++o) {
        if (o == 0) continue;
        int sn = s + o;
        if (sn < 0 || sn >= S_) continue;
        int tn = b * S_ + sn;
        if (mask[tn] != 0) continue;
        cnt += 1.f;
        const unsigned short* Mrow = Mbuf + (size_t)tn * K_ * D_;
#pragma unroll
        for (int kk = 0; kk < K_; ++kk) {
            float wk = wsh[kk];
            const unsigned short* Mk = Mrow + (size_t)kk * D_;
#pragma unroll
            for (int j = 0; j < 8; ++j)
                acc[j] += wk * bf2f(Mk[tid + 256 * j]);
        }
    }

    float inv = 1.f / fmaxf(cnt, 1.f);
    float s1 = 0.f, s2 = 0.f;
#pragma unroll
    for (int j = 0; j < 8; ++j) { acc[j] *= inv; s1 += acc[j]; s2 += acc[j] * acc[j]; }
#pragma unroll
    for (int off = 16; off > 0; off >>= 1) {
        s1 += __shfl_xor(s1, off, 32);
        s2 += __shfl_xor(s2, off, 32);
    }
    if (lane == 0) { red[wave] = s1; red[8 + wave] = s2; }
    __syncthreads();
    if (tid == 0) {
        float t1 = 0.f, t2 = 0.f;
#pragma unroll
        for (int i = 0; i < 8; ++i) { t1 += red[i]; t2 += red[8 + i]; }
        float mu  = t1 / (float)D_;
        float var = t2 / (float)D_ - mu * mu;
        red[0] = mu;
        red[1] = rsqrtf(var + EPS_LN);
    }
    __syncthreads();
    float mu = red[0], rstd = red[1];
    bool  wr = (mask[a] != 0) && (cnt > 0.f);
#pragma unroll
    for (int j = 0; j < 8; ++j) {
        int d = tid + 256 * j;
        float base  = hL[(size_t)a * D_ + d];
        float delta = wr ? ((acc[j] - mu) * rstd) * ALPHA : 0.f;
        out[(size_t)a * D_ + d] = base + delta;
    }
}

// ---------------------------------------------------------------------------
extern "C" void kernel_launch(void* const* d_in, const int* in_sizes, int n_in,
                              void* d_out, int out_size, void* d_ws, size_t ws_size,
                              hipStream_t stream) {
    (void)in_sizes; (void)n_in; (void)out_size; (void)ws_size;
    const float* hL   = (const float*)d_in[0];
    const int*   mask = (const int*)  d_in[1];
    const float* Wr   = (const float*)d_in[2];
    const float* br   = (const float*)d_in[3];
    const float* W1   = (const float*)d_in[4];
    const float* b1   = (const float*)d_in[5];
    const float* W2   = (const float*)d_in[6];
    const float* b2   = (const float*)d_in[7];
    // d_in[8] = range_r, fixed to 8 at compile time
    float* out = (float*)d_out;

    char* ws = (char*)d_ws;
    unsigned short* Mbuf = (unsigned short*)ws;                 // 134,217,728 B
    float*          wbuf = (float*)(ws + 134217728ull);         //     131,072 B
    unsigned short* hp   = (unsigned short*)(ws + 134348800ull);//  20,971,520 B
    unsigned short* w1p  = (unsigned short*)(ws + 155320320ull);//  20,971,520 B
    unsigned short* w2p  = (unsigned short*)(ws + 176291840ull);//  20,971,520 B

    prep_h_kernel <<<1024, 256, 0, stream>>>(hL, hp);
    prep_w1_kernel<<<1024, 256, 0, stream>>>(W1, w1p);
    prep_w2_kernel<<<1024, 256, 0, stream>>>(W2, w2p);

    router_kernel<<<NTOK, 256, 0, stream>>>(hL, Wr, br, wbuf);

    expert_mlp_kernel<<<dim3(NTOK / 32, K_), 256, LDS_TOTAL_BYTES, stream>>>(
        hp, w1p, w2p, b1, b2, Mbuf);

    blend_kernel<<<NTOK, 256, 0, stream>>>(hL, mask, wbuf, Mbuf, out);
}
